// MultiRoundLSHAttention_1314259992892
// MI455X (gfx1250) — matrix operand order repaired and checked by reading, not compiled, not run on hardware
//
#include <hip/hip_runtime.h>
#include <hip/hip_bf16.h>

// ---------------------------------------------------------------------------
// MultiRoundLSHAttention for gfx1250 (MI455X).
// B=1, L=1024, D=512, H=8 (dk=64), R=2 rounds, 64 buckets, chunk=32, K4=64.
// All GEMM-shaped work on v_wmma_f32_16x16x32_bf16 with fragment-swizzled
// operands so every fragment is a contiguous 32B per-lane load (b128 pairs).
// GEMM k-loop is ping-pong software-pipelined (manual unroll-by-2, no moves).
// lsh_scores K-staging uses CDNA5 async global->LDS copies (ASYNCcnt).
// ---------------------------------------------------------------------------

typedef __attribute__((ext_vector_type(8)))  float          v8f;
typedef __attribute__((ext_vector_type(16))) __bf16         v16bf;
typedef __attribute__((ext_vector_type(16))) unsigned short v16us;

#define L_SEQ   1024
#define D_MODEL 512
#define HEADS   8
#define DK      64
#define ROUNDS  2
#define NB2     32      // number of chunks
#define BL2     32      // rows per chunk
#define K4      64      // keys per row (look-back window)
#define BH      8       // B*H

static __device__ inline unsigned short f32_to_bf16(float f) {
  unsigned int b = __builtin_bit_cast(unsigned int, f);
  unsigned int r = (b + 0x7FFFu + ((b >> 16) & 1u)) >> 16;
  return (unsigned short)r;
}

static __device__ inline unsigned int pack2_bf16(float lo, float hi) {
  return (unsigned int)f32_to_bf16(lo) | ((unsigned int)f32_to_bf16(hi) << 16);
}

// ---------------------------------------------------------------------------
// Convert A (f32 M x K) into fragment-swizzled bf16:
//   Asw[((m*(K/32)+g)*2 + h)*16 + e]
// h=0 holds k = g*32 + {0..15}; h=1 holds k = g*32 + {16..31}
// (contiguous k order, the same one the B fragments use; lane-half = h).
// ---------------------------------------------------------------------------
__global__ void conv_a_sw_k(const float* __restrict__ A,
                            unsigned short* __restrict__ Asw, int M, int K) {
  const int KG = K / 32;
  int tid = blockIdx.x * blockDim.x + threadIdx.x;
  if (tid >= M * KG * 2) return;
  const int h = tid & 1;
  const int g = (tid >> 1) % KG;
  const int m = tid / (2 * KG);
  const float* src = A + (size_t)m * K + g * 32;
  unsigned int o[8];
#pragma unroll
  for (int j = 0; j < 4; ++j)
    o[j] = pack2_bf16(src[h * 16 + 2 * j], src[h * 16 + 2 * j + 1]);
#pragma unroll
  for (int j = 0; j < 4; ++j)
    o[4 + j] = pack2_bf16(src[h * 16 + 8 + 2 * j], src[h * 16 + 8 + 2 * j + 1]);
  uint4* dst = (uint4*)(Asw + (size_t)tid * 16);
  dst[0] = uint4{o[0], o[1], o[2], o[3]};
  dst[1] = uint4{o[4], o[5], o[6], o[7]};
}

// ---------------------------------------------------------------------------
// Convert W (f32 K x N) into transposed bf16 Wt[n*K + k]; the B fragment for
// lane (n, kgroup) is then Wt[n*K + kk + kgroup*16 .. +15]: one 32B load.
// ---------------------------------------------------------------------------
__global__ void conv_wt_k(const float* __restrict__ W,
                          unsigned short* __restrict__ Wt, int K, int N) {
  const int KC = K / 16;
  int tid = blockIdx.x * blockDim.x + threadIdx.x;
  if (tid >= N * KC) return;
  const int kc = tid % KC;
  const int n  = tid / KC;
  unsigned int o[8];
#pragma unroll
  for (int j = 0; j < 8; ++j)
    o[j] = pack2_bf16(W[(size_t)(kc * 16 + 2 * j) * N + n],
                      W[(size_t)(kc * 16 + 2 * j + 1) * N + n]);
  uint4* dst = (uint4*)(Wt + (size_t)n * K + kc * 16);
  dst[0] = uint4{o[0], o[1], o[2], o[3]};
  dst[1] = uint4{o[4], o[5], o[6], o[7]};
}

// ---------------------------------------------------------------------------
// C = A*W + bias with WMMA, operands pre-swizzled bf16 in global (L2-resident,
// no LDS needed). Block = 128 threads (4 waves); wave w owns rows
// [16w,16w+16) of a 64x64 tile; 1 A-load + 4 B-loads + 4 WMMAs per 32-k step.
// Ping-pong pipelined: two fragment register sets alternate, loads stay one
// full k-step ahead of the WMMAs with zero register copies. Requires KG even.
// ---------------------------------------------------------------------------
__global__ __launch_bounds__(128) void gemm_bias_wmma(
    const unsigned short* __restrict__ Asw, const unsigned short* __restrict__ Wt,
    const float* __restrict__ bias, float* __restrict__ Out,
    int M, int N, int K) {
  const int t    = threadIdx.x;
  const int lane = t & 31;
  const int w    = t >> 5;
  const int nBase = blockIdx.x * 64;
  const int mBase = blockIdx.y * 64;
  const int KG = K / 32;               // even for all call sites
  const int m  = mBase + w * 16 + (lane & 15);
  const int h  = lane >> 4;

  const unsigned short* arow = Asw + ((size_t)m * KG * 2 + h) * 16;
  const unsigned short* brow[4];
#pragma unroll
  for (int nt = 0; nt < 4; ++nt)
    brow[nt] = Wt + (size_t)(nBase + nt * 16 + (lane & 15)) * K + h * 16;

  const v8f czero = {0.f, 0.f, 0.f, 0.f, 0.f, 0.f, 0.f, 0.f};
  v8f c[4] = {czero, czero, czero, czero};

  v16us a0 = *(const v16us*)(arow);
  v16us b0[4], a1, b1[4];
#pragma unroll
  for (int nt = 0; nt < 4; ++nt) b0[nt] = *(const v16us*)(brow[nt]);

  for (int g = 0; g < KG; g += 2) {
    // loads for step g+1 (always valid: KG even)
    a1 = *(const v16us*)(arow + (g + 1) * 32);
#pragma unroll
    for (int nt = 0; nt < 4; ++nt)
      b1[nt] = *(const v16us*)(brow[nt] + (g + 1) * 32);
    // WMMAs on step g
    {
      v16bf a = __builtin_bit_cast(v16bf, a0);
#pragma unroll
      for (int nt = 0; nt < 4; ++nt)
        c[nt] = __builtin_amdgcn_wmma_f32_16x16x32_bf16(
            false, a, false, __builtin_bit_cast(v16bf, b0[nt]),
            (short)0, c[nt], false, false);
    }
    // loads for step g+2
    if (g + 2 < KG) {
      a0 = *(const v16us*)(arow + (g + 2) * 32);
#pragma unroll
      for (int nt = 0; nt < 4; ++nt)
        b0[nt] = *(const v16us*)(brow[nt] + (g + 2) * 32);
      __builtin_prefetch(arow + (g + 3) * 32, 0, 1);        // global_prefetch_b8
      __builtin_prefetch(brow[0] + (g + 3) * 32, 0, 1);
    }
    // WMMAs on step g+1
    {
      v16bf a = __builtin_bit_cast(v16bf, a1);
#pragma unroll
      for (int nt = 0; nt < 4; ++nt)
        c[nt] = __builtin_amdgcn_wmma_f32_16x16x32_bf16(
            false, a, false, __builtin_bit_cast(v16bf, b1[nt]),
            (short)0, c[nt], false, false);
    }
  }
  // C/D layout: lane<16 -> M=j,N=lane ; lane>=16 -> M=8+j,N=lane-16.
#pragma unroll
  for (int nt = 0; nt < 4; ++nt) {
#pragma unroll
    for (int j = 0; j < 8; ++j) {
      const int mm = w * 16 + (lane >> 4) * 8 + j;
      const int nn = nt * 16 + (lane & 15);
      Out[(size_t)(mBase + mm) * N + nBase + nn] = c[nt][j] + bias[nBase + nn];
    }
  }
}

// ---------------------------------------------------------------------------
// Normalize rand_matrix rows; output layout rmn[((r*32)+k)*DK + d] so the
// hash kernel streams contiguously over d.
// ---------------------------------------------------------------------------
__global__ void rm_norm_k(const float* __restrict__ rm, float* __restrict__ rmn) {
  int t = threadIdx.x;           // (d,r) flattened, rm[(d*R + r)*32 + k]
  if (t >= DK * ROUNDS) return;
  const int d = t >> 1, r = t & 1;
  const float* src = rm + (d * ROUNDS + r) * 32;
  float s = 0.f;
  for (int k = 0; k < 32; ++k) s += src[k] * src[k];
  float inv = rsqrtf(s);
  for (int k = 0; k < 32; ++k) rmn[(r * 32 + k) * DK + d] = src[k] * inv;
}

// ---------------------------------------------------------------------------
// Per (l,h) row: emit bf16 q and bf16 k = q/||q|| in [bh][l][d] layout.
// ---------------------------------------------------------------------------
__global__ void qk_prep_k(const float* __restrict__ q2d,
                          unsigned short* __restrict__ qbf,
                          unsigned short* __restrict__ kbf) {
  int tid = blockIdx.x * blockDim.x + threadIdx.x;   // L*H rows
  if (tid >= L_SEQ * HEADS) return;
  int h = tid & (HEADS - 1);
  int l = tid >> 3;
  const float* q = q2d + (size_t)l * D_MODEL + h * DK;
  float s = 0.f;
  for (int d = 0; d < DK; ++d) s += q[d] * q[d];
  float inv = rsqrtf(s);
  unsigned short* qo = qbf + (size_t)(h * L_SEQ + l) * DK;
  unsigned short* ko = kbf + (size_t)(h * L_SEQ + l) * DK;
  for (int d = 0; d < DK; ++d) {
    qo[d] = f32_to_bf16(q[d]);
    ko[d] = f32_to_bf16(q[d] * inv);
  }
}

// ---------------------------------------------------------------------------
// LSH hash: argmax over [proj, -proj] (64 logical buckets) per (bh,l,r).
// ---------------------------------------------------------------------------
__global__ void lsh_hash_k(const float* __restrict__ q2d,
                           const float* __restrict__ rmn,
                           int* __restrict__ hashes) {
  int tid = blockIdx.x * blockDim.x + threadIdx.x;   // BH*L*R
  if (tid >= BH * L_SEQ * ROUNDS) return;
  int r  = tid & 1;
  int l  = (tid >> 1) & (L_SEQ - 1);
  int bh = tid >> 11;
  const float* q = q2d + (size_t)l * D_MODEL + bh * DK;
  float best = -1e30f;
  int bi = 0;
  for (int k = 0; k < 32; ++k) {
    const float* rv = rmn + (r * 32 + k) * DK;
    float acc = 0.f;
    for (int d = 0; d < DK; ++d) acc += q[d] * rv[d];
    if (acc > best)  { best = acc;  bi = k; }
    if (-acc > best) { best = -acc; bi = k + 32; }
  }
  hashes[(bh * ROUNDS + r) * L_SEQ + l] = bi;
}

// ---------------------------------------------------------------------------
// Stable counting sort per (bh,r): hash_idx = argsort(hashes), orig_idx = inv.
// ---------------------------------------------------------------------------
__global__ __launch_bounds__(64) void lsh_sort_k(const int* __restrict__ hashes,
                                                 int* __restrict__ hash_idx,
                                                 int* __restrict__ orig_idx) {
  int br = blockIdx.x;                       // bh*R + r
  const int* h = hashes + br * L_SEQ;
  __shared__ int cnt[64];
  __shared__ int off[64];
  int b = threadIdx.x;
  int c = 0;
  for (int l = 0; l < L_SEQ; ++l) c += (h[l] == b);
  cnt[b] = c;
  __syncthreads();
  if (b == 0) {
    int s = 0;
    for (int i = 0; i < 64; ++i) { off[i] = s; s += cnt[i]; }
  }
  __syncthreads();
  int o = off[b];
  for (int l = 0; l < L_SEQ; ++l) {
    if (h[l] == b) {
      hash_idx[br * L_SEQ + o] = l;
      orig_idx[br * L_SEQ + l] = o;
      ++o;
    }
  }
}

// ---------------------------------------------------------------------------
// Block attention scores: per (bh, chunk n, round r): S = Q.K^T/8 via WMMA.
// Gathered rows staged into LDS in fragment-swizzled order so fragment reads
// are contiguous 32B (ds_load_b128 pairs). K-rows are staged with the CDNA5
// async global->LDS DMA path (GLOBAL_LOAD_ASYNC_TO_LDS_B128 / ASYNCcnt).
//   qsw fragment f = kkIdx*2 + mt (4 frags), ksw fragment f = kkIdx*4 + nt (8).
// ---------------------------------------------------------------------------
__global__ __launch_bounds__(128) void lsh_scores_k(
    const unsigned short* __restrict__ qbf, const unsigned short* __restrict__ kbf,
    const int* __restrict__ hashes, const int* __restrict__ hash_idx,
    float* __restrict__ scores) {
  const int idx = blockIdx.x;           // ((bh*NB2 + n)*ROUNDS + r)
  const int r   = idx & 1;
  const int n   = (idx >> 1) & (NB2 - 1);
  const int bh  = idx >> 6;
  const int br  = bh * ROUNDS + r;

  __shared__ __align__(32) unsigned short qsw[4 * 32 * 16];
  __shared__ __align__(32) unsigned short ksw[8 * 32 * 16];
  __shared__ int   lq[BL2], hq[BL2];
  __shared__ int   lk[K4],  hk[K4];
  __shared__ float clog[K4];
  __shared__ int   kval[K4];

  const int t = threadIdx.x;
  if (t < 32) {
    int p = n * BL2 + t;
    int l = hash_idx[br * L_SEQ + p];
    lq[t] = l;
    hq[t] = hashes[br * L_SEQ + l];
  } else if (t < 96) {
    int j = t - 32;
    int kp = (n - 1) * BL2 + j;
    int valid = (kp >= 0);
    int l = valid ? hash_idx[br * L_SEQ + kp] : 0;   // ref pads rk with 0
    lk[j] = l;
    hk[j] = valid ? hashes[br * L_SEQ + l] : -1;
    kval[j] = valid;
  }
  __syncthreads();

  // Stage Q rows: 32 rows x 8 chunks of 8 bf16 (16B each).
  for (int i = t; i < 256; i += 128) {
    const int mrow  = i >> 3;
    const int cch   = i & 7;            // k = cch*8 .. cch*8+7
    const int kkIdx = cch >> 2;
    const int rel   = (cch & 3) * 8;    // 0,8,16,24
    const int kgrp  = rel >> 4;
    const int half  = (rel >> 3) & 1;
    const int mt    = mrow >> 4;
    const int lanei = kgrp * 16 + (mrow & 15);
    const unsigned short* src = qbf + (size_t)(bh * L_SEQ + lq[mrow]) * DK + cch * 8;
    unsigned short* dst = &qsw[((kkIdx * 2 + mt) * 32 + lanei) * 16 + half * 8];
    *(uint4*)dst = *(const uint4*)src;
  }
  // Stage K rows: 64 keys x 4 chunks of 16 bf16 (32B each), via async DMA to
  // LDS: per-lane global address -> per-lane LDS address, tracked by ASYNCcnt.
  for (int i = t; i < 256; i += 128) {
    const int nrow  = i >> 2;
    const int cch   = i & 3;            // k = cch*16 .. +15
    const int kkIdx = cch >> 1;
    const int kgrp  = cch & 1;
    const int nt    = nrow >> 4;
    const int lanei = kgrp * 16 + (nrow & 15);
    unsigned short* dst = &ksw[((kkIdx * 4 + nt) * 32 + lanei) * 16];
    if (kval[nrow]) {
      const unsigned short* src = kbf + (size_t)(bh * L_SEQ + lk[nrow]) * DK + cch * 16;
      const unsigned ldsOff0 = (unsigned)(size_t)(dst);
      const unsigned ldsOff1 = (unsigned)(size_t)(dst + 8);
      asm volatile("global_load_async_to_lds_b128 %0, %1, off"
                   :: "v"(ldsOff0), "v"(src) : "memory");
      asm volatile("global_load_async_to_lds_b128 %0, %1, off"
                   :: "v"(ldsOff1), "v"(src + 8) : "memory");
    } else {
      uint4 z = {0u, 0u, 0u, 0u};
      *(uint4*)(dst) = z;
      *(uint4*)(dst + 8) = z;
    }
  }
  asm volatile("s_wait_asynccnt 0" ::: "memory");   // drain async DMA
  if (t < 64) {
    int c = 0;
    for (int j = 0; j < K4; ++j) c += (lk[j] == lk[t]);
    clog[t] = logf((float)c);
  }
  __syncthreads();

  const int w = t >> 5, lane = t & 31;
  const v8f czero = {0.f, 0.f, 0.f, 0.f, 0.f, 0.f, 0.f, 0.f};
  v8f c0 = czero, c1 = czero;   // M-tiles: rows 0..15 and 16..31
#pragma unroll
  for (int kkIdx = 0; kkIdx < 2; ++kkIdx) {
    v16bf a0 = __builtin_bit_cast(v16bf, *(const v16us*)&qsw[((kkIdx * 2 + 0) * 32 + lane) * 16]);
    v16bf a1 = __builtin_bit_cast(v16bf, *(const v16us*)&qsw[((kkIdx * 2 + 1) * 32 + lane) * 16]);
    v16bf b  = __builtin_bit_cast(v16bf, *(const v16us*)&ksw[((kkIdx * 4 + w) * 32 + lane) * 16]);
    c0 = __builtin_amdgcn_wmma_f32_16x16x32_bf16(false, a0, false, b, (short)0, c0, false, false);
    c1 = __builtin_amdgcn_wmma_f32_16x16x32_bf16(false, a1, false, b, (short)0, c1, false, false);
  }

  float* out = scores + (size_t)(br * L_SEQ + n * BL2) * K4;
#pragma unroll
  for (int j = 0; j < 8; ++j) {
    const int nn = w * 16 + (lane & 15);
#pragma unroll
    for (int mt = 0; mt < 2; ++mt) {
      const int m = mt * 16 + (lane >> 4) * 8 + j;
      float s = (mt ? c1[j] : c0[j]) * 0.125f;        // / sqrt(64)
      if (!kval[nn] || hq[m] != hk[nn] || lq[m] <= lk[nn]) s -= 1e9f;
      if (lq[m] == lk[nn]) s -= 1e5f;
      s -= clog[nn];
      out[m * K4 + nn] = s;
    }
  }
}

// ---------------------------------------------------------------------------
// Softmax over both rounds (128 logits) per (bh,l) and V gather-accumulate.
// Wave32 butterfly reductions via __shfl_xor.
// ---------------------------------------------------------------------------
__global__ __launch_bounds__(32) void lsh_av_k(
    const float* __restrict__ scores, const float* __restrict__ v2d,
    const int* __restrict__ hash_idx, const int* __restrict__ orig_idx,
    float* __restrict__ attn) {
  const int bid = blockIdx.x;            // bh*L + l
  const int l   = bid & (L_SEQ - 1);
  const int bh  = bid >> 10;
  __shared__ float wgt[2 * K4];
  __shared__ int   ko[2 * K4];
  const int t = threadIdx.x;
  for (int j = t; j < 2 * K4; j += 32) {
    int r  = j >> 6, jj = j & 63;
    int br = bh * ROUNDS + r;
    int p  = orig_idx[br * L_SEQ + l];
    int nc = p >> 5;
    int kp = (nc - 1) * BL2 + jj;
    ko[j]  = (kp >= 0) ? hash_idx[br * L_SEQ + kp] : -1;
    wgt[j] = scores[(size_t)(br * L_SEQ + p) * K4 + jj];
  }
  __syncthreads();
  float lmax = -1e30f;
  for (int j = t; j < 2 * K4; j += 32) lmax = fmaxf(lmax, wgt[j]);
#pragma unroll
  for (int o = 16; o > 0; o >>= 1) lmax = fmaxf(lmax, __shfl_xor(lmax, o, 32));
  float lsum = 0.f;
  for (int j = t; j < 2 * K4; j += 32) {
    float e = __expf(wgt[j] - lmax);
    wgt[j] = e;
    lsum += e;
  }
#pragma unroll
  for (int o = 16; o > 0; o >>= 1) lsum += __shfl_xor(lsum, o, 32);
  __syncthreads();
  const float inv = 1.0f / lsum;
  float acc0 = 0.f, acc1 = 0.f;
  const int d0 = t, d1 = t + 32;
  for (int j = 0; j < 2 * K4; ++j) {
    int k = ko[j];
    if (k < 0) continue;
    float p = wgt[j] * inv;
    const float* vr = v2d + (size_t)k * D_MODEL + bh * DK;
    acc0 += p * vr[d0];
    acc1 += p * vr[d1];
  }
  attn[(size_t)l * D_MODEL + bh * DK + d0] = acc0;
  attn[(size_t)l * D_MODEL + bh * DK + d1] = acc1;
}

// ---------------------------------------------------------------------------
extern "C" void kernel_launch(void* const* d_in, const int* in_sizes, int n_in,
                              void* d_out, int out_size, void* d_ws, size_t ws_size,
                              hipStream_t stream) {
  (void)in_sizes; (void)n_in; (void)out_size; (void)ws_size;
  const float* query = (const float*)d_in[0];
  // d_in[1] = key (unused by reference; k derived from q), d_in[3] = mask (all true)
  const float* value = (const float*)d_in[2];
  const float* Wq = (const float*)d_in[4];
  const float* bq = (const float*)d_in[5];
  const float* Wv = (const float*)d_in[6];
  const float* bv = (const float*)d_in[7];
  const float* Wo = (const float*)d_in[8];
  const float* bo = (const float*)d_in[9];
  const float* rm = (const float*)d_in[10];

  char* ws = (char*)d_ws;
  size_t off = 0;
  auto alloc = [&](size_t bytes) {
    void* p = ws + off;
    off = (off + bytes + 255) & ~size_t(255);
    return p;
  };
  float* q2d  = (float*)alloc((size_t)L_SEQ * D_MODEL * sizeof(float));
  float* v2d  = (float*)alloc((size_t)L_SEQ * D_MODEL * sizeof(float));
  unsigned short* qbf = (unsigned short*)alloc((size_t)BH * L_SEQ * DK * sizeof(unsigned short));
  unsigned short* kbf = (unsigned short*)alloc((size_t)BH * L_SEQ * DK * sizeof(unsigned short));
  float* rmn  = (float*)alloc((size_t)DK * ROUNDS * 32 * sizeof(float));
  int* hashes   = (int*)alloc((size_t)BH * ROUNDS * L_SEQ * sizeof(int));
  int* hash_idx = (int*)alloc((size_t)BH * ROUNDS * L_SEQ * sizeof(int));
  int* orig_idx = (int*)alloc((size_t)BH * ROUNDS * L_SEQ * sizeof(int));
  float* sc   = (float*)alloc((size_t)BH * ROUNDS * L_SEQ * K4 * sizeof(float));
  float* attn = (float*)alloc((size_t)L_SEQ * D_MODEL * sizeof(float));
  unsigned short* Asw = (unsigned short*)alloc((size_t)L_SEQ * D_MODEL * sizeof(unsigned short));
  unsigned short* Wt  = (unsigned short*)alloc((size_t)D_MODEL * D_MODEL * sizeof(unsigned short));

  const int M = L_SEQ, N = D_MODEL, K = D_MODEL;
  dim3 ggrid(N / 64, M / 64);
  const int convA_tasks  = M * (K / 32) * 2;
  const int convWt_tasks = N * (K / 16);

  // q = query*Wq + bq
  conv_a_sw_k<<<(convA_tasks + 255) / 256, 256, 0, stream>>>(query, Asw, M, K);
  conv_wt_k<<<(convWt_tasks + 255) / 256, 256, 0, stream>>>(Wq, Wt, K, N);
  gemm_bias_wmma<<<ggrid, 128, 0, stream>>>(Asw, Wt, bq, q2d, M, N, K);
  // v = value*Wv + bv
  conv_a_sw_k<<<(convA_tasks + 255) / 256, 256, 0, stream>>>(value, Asw, M, K);
  conv_wt_k<<<(convWt_tasks + 255) / 256, 256, 0, stream>>>(Wv, Wt, K, N);
  gemm_bias_wmma<<<ggrid, 128, 0, stream>>>(Asw, Wt, bv, v2d, M, N, K);

  rm_norm_k<<<1, 128, 0, stream>>>(rm, rmn);
  qk_prep_k<<<(L_SEQ * HEADS + 255) / 256, 256, 0, stream>>>(q2d, qbf, kbf);
  lsh_hash_k<<<(BH * L_SEQ * ROUNDS + 255) / 256, 256, 0, stream>>>(q2d, rmn, hashes);
  lsh_sort_k<<<BH * ROUNDS, 64, 0, stream>>>(hashes, hash_idx, orig_idx);
  lsh_scores_k<<<BH * NB2 * ROUNDS, 128, 0, stream>>>(qbf, kbf, hashes, hash_idx, sc);
  lsh_av_k<<<BH * L_SEQ, 32, 0, stream>>>(sc, v2d, hash_idx, orig_idx, attn);

  // out = attn*Wo + bo
  conv_a_sw_k<<<(convA_tasks + 255) / 256, 256, 0, stream>>>(attn, Asw, M, K);
  conv_wt_k<<<(convWt_tasks + 255) / 256, 256, 0, stream>>>(Wo, Wt, K, N);
  gemm_bias_wmma<<<ggrid, 128, 0, stream>>>(Asw, Wt, bo, (float*)d_out, M, N, K);
}
